// RGCN_48000554500364
// MI455X (gfx1250) — compile-verified, matrix operand order checked by reading, not run on hardware
//
#include <hip/hip_runtime.h>
#include <hip/hip_bf16.h>
#include <math.h>

#define N_NODES 50000
#define IN_F    128
#define HID     128
#define N_REL   8
#define MTILES  (N_NODES / 16)    // 3125, exact
#define NMATS   (N_REL + 1)       // 8 relations + self-loop
#define GROUPS  ((MTILES + 7) / 8)  // 391 groups of 8 mtiles (one per wave)

typedef __bf16 v16bf __attribute__((ext_vector_type(16)));
typedef float  v8f   __attribute__((ext_vector_type(8)));
typedef int    v4i   __attribute__((vector_size(16)));

typedef __attribute__((address_space(1))) v4i as1_v4i;   // global
typedef __attribute__((address_space(3))) v4i as3_v4i;   // LDS

union ABPack {
    v16bf v;
    uint4 q[2];
};

#define HAS_ASYNC_LDS  __has_builtin(__builtin_amdgcn_global_load_async_to_lds_b128)
#define HAS_WAIT_ASYNC __has_builtin(__builtin_amdgcn_s_wait_asynccnt)

// ---------------------------------------------------------------------------
// f32 -> bf16 conversion / ReLU+convert
// ---------------------------------------------------------------------------
__global__ void k_cvt_bf16(const float* __restrict__ in, __bf16* __restrict__ out, int count) {
    int i = blockIdx.x * blockDim.x + threadIdx.x;
    if (i < count) out[i] = (__bf16)in[i];
}

__global__ void k_relu_bf16(const float* __restrict__ in, __bf16* __restrict__ out, int count) {
    int i = blockIdx.x * blockDim.x + threadIdx.x;
    if (i < count) {
        float v = in[i];
        out[i] = (__bf16)(v > 0.0f ? v : 0.0f);
    }
}

// ---------------------------------------------------------------------------
// Pre-swizzle weights into the exact WMMA B-operand lane layout (bf16).
// Bsw element index = (((mat*4 + kk)*8 + ntile)*32 + lane)*16 + e
//   e in [0,16): K_local = (e>>3)*16 + (lane>>4)*8 + (e&7)   (ISA 16-bit B layout)
//   k = kk*32 + K_local ;  n = ntile*16 + (lane&15)
// mat 0..7 = W[rel], mat 8 = W_self.
// ---------------------------------------------------------------------------
__global__ void k_pack_B(const float* __restrict__ W, const float* __restrict__ W_self,
                         __bf16* __restrict__ Bsw) {
    int idx = blockIdx.x * blockDim.x + threadIdx.x;
    const int total = NMATS * 4 * 8 * 32 * 16;  // 147456
    if (idx >= total) return;
    int e     = idx & 15;
    int lane  = (idx >> 4) & 31;
    int ntile = (idx >> 9) & 7;
    int kk    = (idx >> 12) & 3;
    int mat   = idx >> 14;
    int klocal = ((e >> 3) << 4) + ((lane >> 4) << 3) + (e & 7);
    int k = kk * 32 + klocal;
    int n = (ntile << 4) + (lane & 15);
    float w = (mat < N_REL) ? W[((size_t)mat * IN_F + k) * HID + n]
                            : W_self[(size_t)k * HID + n];
    Bsw[idx] = (__bf16)w;
}

// ---------------------------------------------------------------------------
// Fused 9-matrix GEMM with LDS-staged B operand.
// Block = (mat, group of 8 mtiles). The whole workgroup async-copies the
// 32 KB swizzled weight matrix into LDS once (GLOBAL_LOAD_ASYNC_TO_LDS_B128,
// ASYNCcnt), then each wave computes a 16x128 output strip:
//   mat 0..7 : xr[mat] = h @ W[mat]          (written f32)
//   mat 8    : agg     = h @ W_self + bias   (seeds the aggregation buffer)
// ---------------------------------------------------------------------------
__global__ void __launch_bounds__(256)
k_rgcn_gemm(const __bf16* __restrict__ hbf, const __bf16* __restrict__ Bsw,
            const float* __restrict__ bias,
            float* __restrict__ xr, float* __restrict__ agg) {
    const int mat  = blockIdx.x / GROUPS;       // 0..8
    const int grp  = blockIdx.x % GROUPS;
    const int widx = threadIdx.x >> 5;          // wave in block (0..7)
    const int lane = threadIdx.x & 31;
    const int mtile = grp * 8 + widx;

    __shared__ __align__(16) uint4 Bsh[2048];   // 32 KB: full swizzled matrix

    // --- cooperative B staging: 256 threads x 8 x 16B = 32 KB ---
    {
        const uint4* gB = (const uint4*)(Bsw + (size_t)mat * (4 * 8 * 32 * 16));
        int t = threadIdx.x;
#if HAS_ASYNC_LDS && HAS_WAIT_ASYNC
        as1_v4i* gsrc = (as1_v4i*)(gB + t);
        as3_v4i* ldst = (as3_v4i*)(Bsh + t);
#pragma unroll
        for (int i = 0; i < 8; ++i)
            __builtin_amdgcn_global_load_async_to_lds_b128(gsrc + i * 256, ldst + i * 256, 0, 0);
        __builtin_amdgcn_s_wait_asynccnt(0);
#else
#pragma unroll
        for (int i = 0; i < 8; ++i)
            Bsh[t + i * 256] = gB[t + i * 256];
#endif
    }
    __syncthreads();
    if (mtile >= MTILES) return;   // after the barrier: all threads participate above

    const int hi = lane >> 4;      // K-half selector
    const int lo = lane & 15;      // row (A) / column (B,D) within tile
    const int m  = mtile * 16 + lo;

    const v8f zero = {0.f, 0.f, 0.f, 0.f, 0.f, 0.f, 0.f, 0.f};
    v8f acc[8];
#pragma unroll
    for (int i = 0; i < 8; ++i) acc[i] = zero;

    const uint4* __restrict__ arow = (const uint4*)(hbf + (size_t)m * IN_F);

#pragma unroll
    for (int kk = 0; kk < 4; ++kk) {
        ABPack a;
        // element offsets kk*32 + g*16 + hi*8  ->  uint4 index kk*4 + g*2 + hi
        a.q[0] = arow[kk * 4 + hi];
        a.q[1] = arow[kk * 4 + 2 + hi];
#pragma unroll
        for (int nt = 0; nt < 8; ++nt) {
            ABPack b;
            int bi = ((kk * 8 + nt) * 32 + lane) * 2;   // uint4 index into LDS
            b.q[0] = Bsh[bi];
            b.q[1] = Bsh[bi + 1];
            acc[nt] = __builtin_amdgcn_wmma_f32_16x16x32_bf16(
                false, a.v, false, b.v, (short)0, acc[nt], false, false);
        }
    }

    // Epilogue: D layout — VGPR v holds row (v + 8*hi), column lo.
    if (mat < N_REL) {
        float* out = xr + (size_t)mat * N_NODES * HID;
#pragma unroll
        for (int nt = 0; nt < 8; ++nt) {
            int col = nt * 16 + lo;
#pragma unroll
            for (int v = 0; v < 8; ++v) {
                int row = mtile * 16 + v + hi * 8;
                out[(size_t)row * HID + col] = acc[nt][v];
            }
        }
    } else {
#pragma unroll
        for (int nt = 0; nt < 8; ++nt) {
            int col = nt * 16 + lo;
            float bcol = bias[col];
#pragma unroll
            for (int v = 0; v < 8; ++v) {
                int row = mtile * 16 + v + hi * 8;
                agg[(size_t)row * HID + col] = acc[nt][v] + bcol;
            }
        }
    }
}

// ---------------------------------------------------------------------------
// Edge scatter: one wave per edge, each lane moves 4 floats (32*4 = 128).
// agg[dst] += xr[etype][src]   via global_atomic_add_f32.
// ---------------------------------------------------------------------------
__global__ void k_edge_scatter(const float* __restrict__ xr,
                               const int* __restrict__ src, const int* __restrict__ dst,
                               const int* __restrict__ et, float* __restrict__ agg,
                               int n_edges) {
    int wid  = (int)((blockIdx.x * blockDim.x + threadIdx.x) >> 5);
    int lane = threadIdx.x & 31;
    if (wid >= n_edges) return;
    int s = src[wid];
    int d = dst[wid];
    int r = et[wid];
    const float4* mrow = (const float4*)(xr + ((size_t)r * N_NODES + s) * HID);
    float4 msg = mrow[lane];
    float* arow = agg + (size_t)d * HID + lane * 4;
    atomicAdd(arow + 0, msg.x);
    atomicAdd(arow + 1, msg.y);
    atomicAdd(arow + 2, msg.z);
    atomicAdd(arow + 3, msg.w);
}

// ---------------------------------------------------------------------------
// Pooling: relu then column-wise partial sums into pooled[128] (atomics).
// ---------------------------------------------------------------------------
__global__ void k_zero_pool(float* pooled) {
    pooled[threadIdx.x] = 0.0f;
}

__global__ void k_relu_pool(const float* __restrict__ agg, float* __restrict__ pooled) {
    int col  = threadIdx.x;          // 128 threads
    int row0 = blockIdx.x * 256;
    float s = 0.0f;
    for (int r = 0; r < 256; ++r) {
        int row = row0 + r;
        if (row >= N_NODES) break;
        float v = agg[(size_t)row * HID + col];
        s += (v > 0.0f ? v : 0.0f);
    }
    atomicAdd(&pooled[col], s);
}

// ---------------------------------------------------------------------------
// Final: mean, fc [128x1], sigmoid.
// ---------------------------------------------------------------------------
__global__ void k_final(const float* __restrict__ pooled, const float* __restrict__ fc_w,
                        const float* __restrict__ fc_b, float* __restrict__ out) {
    __shared__ float sh[128];
    int t = threadIdx.x;
    sh[t] = pooled[t] * (1.0f / (float)N_NODES) * fc_w[t];
    __syncthreads();
    for (int s = 64; s > 0; s >>= 1) {
        if (t < s) sh[t] += sh[t + s];
        __syncthreads();
    }
    if (t == 0) {
        float z = sh[0] + fc_b[0];
        out[0] = 1.0f / (1.0f + expf(-z));
    }
}

// ---------------------------------------------------------------------------
extern "C" void kernel_launch(void* const* d_in, const int* in_sizes, int n_in,
                              void* d_out, int out_size, void* d_ws, size_t ws_size,
                              hipStream_t stream) {
    const float* in_feat = (const float*)d_in[0];
    const float* W1      = (const float*)d_in[1];
    const float* W1_self = (const float*)d_in[2];
    const float* b1      = (const float*)d_in[3];
    const float* W2      = (const float*)d_in[4];
    const float* W2_self = (const float*)d_in[5];
    const float* b2      = (const float*)d_in[6];
    const float* fc_w    = (const float*)d_in[7];
    const float* fc_b    = (const float*)d_in[8];
    const int*   src     = (const int*)d_in[9];
    const int*   dst     = (const int*)d_in[10];
    const int*   etypes  = (const int*)d_in[11];
    float* out = (float*)d_out;
    const int n_edges = in_sizes[9];

    // Workspace carve-out (256B aligned).
    char* ws = (char*)d_ws;
    size_t off = 0;
    auto alloc = [&](size_t bytes) -> void* {
        off = (off + 255) & ~(size_t)255;
        void* p = ws + off;
        off += bytes;
        return p;
    };
    __bf16* hbf   = (__bf16*)alloc((size_t)N_NODES * IN_F * sizeof(__bf16));       // 12.8 MB
    __bf16* Bsw1  = (__bf16*)alloc((size_t)NMATS * 4 * 8 * 32 * 16 * sizeof(__bf16));
    __bf16* Bsw2  = (__bf16*)alloc((size_t)NMATS * 4 * 8 * 32 * 16 * sizeof(__bf16));
    float*  xr    = (float*)alloc((size_t)N_REL * N_NODES * HID * sizeof(float));  // 204.8 MB
    float*  agg   = (float*)alloc((size_t)N_NODES * HID * sizeof(float));          // 25.6 MB
    float*  pooled= (float*)alloc(128 * sizeof(float));

    const int elems = N_NODES * HID;                 // 6.4M
    const int cvt_blocks  = (elems + 255) / 256;
    const int pack_blocks = (NMATS * 4 * 8 * 32 * 16 + 255) / 256;
    const int gemm_blocks = NMATS * GROUPS;          // 9 * 391 = 3519 blocks
    const int edge_blocks = (n_edges + 7) / 8;       // 8 edges (waves) / block
    const int pool_blocks = (N_NODES + 255) / 256;

    // Weights -> WMMA-swizzled bf16 (once per launch, deterministic).
    k_pack_B<<<pack_blocks, 256, 0, stream>>>(W1, W1_self, Bsw1);
    k_pack_B<<<pack_blocks, 256, 0, stream>>>(W2, W2_self, Bsw2);

    // ---- Layer 1 ----
    k_cvt_bf16<<<cvt_blocks, 256, 0, stream>>>(in_feat, hbf, elems);
    k_rgcn_gemm<<<gemm_blocks, 256, 0, stream>>>(hbf, Bsw1, b1, xr, agg);
    k_edge_scatter<<<edge_blocks, 256, 0, stream>>>(xr, src, dst, etypes, agg, n_edges);
    k_relu_bf16<<<cvt_blocks, 256, 0, stream>>>(agg, hbf, elems);   // h1 (bf16)

    // ---- Layer 2 ----
    k_rgcn_gemm<<<gemm_blocks, 256, 0, stream>>>(hbf, Bsw2, b2, xr, agg);
    k_edge_scatter<<<edge_blocks, 256, 0, stream>>>(xr, src, dst, etypes, agg, n_edges);

    // ---- Pool + FC + sigmoid ----
    k_zero_pool<<<1, 128, 0, stream>>>(pooled);
    k_relu_pool<<<pool_blocks, 128, 0, stream>>>(agg, pooled);
    k_final<<<1, 128, 0, stream>>>(pooled, fc_w, fc_b, out);
}
